// Multi_head_54133767799373
// MI455X (gfx1250) — matrix operand order repaired and checked by reading, not compiled, not run on hardware
//
#include <hip/hip_runtime.h>

#define D      128
#define NE     5
#define BROWS  262144
#define MTILE  128
#define NTPE   (BROWS / MTILE)       // 2048 tiles per expert (max)
#define WSZ    (NE * D * D)          // 81920 elements per weight tensor

typedef __attribute__((ext_vector_type(16))) _Float16 v16h;
typedef __attribute__((ext_vector_type(8)))  _Float16 v8h;
typedef __attribute__((ext_vector_type(8)))  float    v8f;

__device__ __forceinline__ int expert_of(float t) {
    const float PT[6] = {0.0f, 0.2f, 0.4f, 0.6f, 0.8f, 1.0f};
    int e = -1;
#pragma unroll
    for (int i = 0; i < NE; i++)
        if (t >= PT[i] && t < PT[i + 1]) e = i;
    return e;
}

// ---------------- routing ----------------
__global__ void route_count(const float* __restrict__ x, int* __restrict__ counts) {
    int r = blockIdx.x * blockDim.x + threadIdx.x;
    if (r < BROWS) {
        int e = expert_of(x[(size_t)r * (D + 1)]);
        if (e >= 0) atomicAdd(&counts[e], 1);
    }
}

__global__ void route_scan(const int* __restrict__ counts, int* __restrict__ offsets,
                           int* __restrict__ cursor) {
    if (threadIdx.x == 0 && blockIdx.x == 0) {
        int acc = 0;
        for (int i = 0; i < NE; i++) {
            offsets[i] = acc;
            cursor[i]  = acc;
            acc += counts[i];
        }
        offsets[NE] = acc;
    }
}

__global__ void route_scatter(const float* __restrict__ x, int* __restrict__ cursor,
                              int* __restrict__ idx) {
    int r = blockIdx.x * blockDim.x + threadIdx.x;
    if (r < BROWS) {
        int e = expert_of(x[(size_t)r * (D + 1)]);
        if (e >= 0) {
            int p  = atomicAdd(&cursor[e], 1);
            idx[p] = r;
        }
    }
}

// ------------- weight prep: f32 -> f16 in WMMA B-fragment layout -------------
// Fragment (kt,nt): lane l (0..15): N = nt*16 + l, halves j: K = kt*32 + j
//                   lane l+16     : N = nt*16 + l, halves j: K = kt*32 + 16 + j
__global__ void prep_weights(const float* __restrict__ W1, const float* __restrict__ W2,
                             _Float16* __restrict__ W1p, _Float16* __restrict__ W2p) {
    int tid = blockIdx.x * blockDim.x + threadIdx.x;     // 0 .. 2*WSZ-1
    int which = tid / WSZ;
    int rem   = tid % WSZ;
    int e     = rem / (D * D);
    int f     = rem % (D * D);
    int kt    = f >> 12;            // /4096
    int nt    = (f >> 9) & 7;
    int lane  = (f >> 4) & 31;
    int j     = f & 15;
    int k     = kt * 32 + ((lane & 16) ? 16 : 0) + j;
    int n     = nt * 16 + (lane & 15);
    const float* src = which ? W2 : W1;
    _Float16*    dst = which ? W2p : W1p;
    dst[e * (D * D) + f] = (_Float16)src[e * (D * D) + k * D + n];
}

// ---------------- main fused MoE-MLP kernel ----------------
__global__ __launch_bounds__(256, 1) void moe_mlp(
    const float* __restrict__ x,
    const _Float16* __restrict__ W1p, const _Float16* __restrict__ W2p,
    const float* __restrict__ tw1, const float* __restrict__ b1,
    const float* __restrict__ tw2, const float* __restrict__ b2,
    const float* __restrict__ W3,  const float* __restrict__ tw3,
    const float* __restrict__ b3,
    const int* __restrict__ offsets, const int* __restrict__ idx,
    float* __restrict__ out)
{
    __shared__ __align__(16) _Float16 lds_A[MTILE][D + 8];  // +8 halfs: bank-conflict pad
    __shared__ __align__(16) _Float16 lds_H[MTILE][D + 8];
    __shared__ __align__(16) _Float16 lds_W[4 * 8 * 32 * 16]; // 32 KB fragment store
    __shared__ float lds_t[MTILE];
    __shared__ float lds_w3[D];
    __shared__ int   lds_row[MTILE];

    const int e    = blockIdx.x / NTPE;
    const int tile = blockIdx.x % NTPE;
    const int base = offsets[e] + tile * MTILE;
    const int end  = offsets[e + 1];
    if (base >= end) return;
    const int nrows = min(MTILE, end - base);

    const int tid  = threadIdx.x;
    const int lane = tid & 31;
    const int w    = tid >> 5;

    // ---- gather 128 rows of f (f32 -> f16) + t ----
    {
        int row  = tid >> 1;          // 0..127
        int half = tid & 1;           // 64 features each
        int gRow = (row < nrows) ? idx[base + row] : -1;
        if (half == 0) {
            lds_row[row] = gRow;
            lds_t[row]   = (gRow >= 0) ? x[(size_t)gRow * (D + 1)] : 0.0f;
        }
        if (gRow >= 0) {
            const float* src = x + (size_t)gRow * (D + 1) + 1 + half * 64;
#pragma unroll
            for (int c = 0; c < 8; c++) {
                v8h pk;
#pragma unroll
                for (int q = 0; q < 8; q++) pk[q] = (_Float16)src[c * 8 + q];
                *(v8h*)&lds_A[row][half * 64 + c * 8] = pk;
            }
        } else {
#pragma unroll
            for (int c = 0; c < 8; c++) {
                v8h z = {};
                *(v8h*)&lds_A[row][half * 64 + c * 8] = z;
            }
        }
        if (tid < D) lds_w3[tid] = W3[e * D + tid];
    }
    // ---- stage W1 fragments ----
    {
        const uint4* src = (const uint4*)(W1p + e * (D * D));
        uint4*       dst = (uint4*)lds_W;
#pragma unroll
        for (int c = 0; c < 8; c++) dst[tid + c * 256] = src[tid + c * 256];
    }
    __syncthreads();

    // per-lane t values for the 8 C rows this lane owns
    float tv[8];
#pragma unroll
    for (int j = 0; j < 8; j++) {
        int m  = (lane < 16) ? j : (8 + j);
        tv[j]  = lds_t[16 * w + m];
    }
    const int nlow  = lane & 15;
    const int arow  = 16 * w + nlow;          // A-fragment row for this lane
    const int k0off = (lane & 16) ? 16 : 0;   // A-fragment K sub-offset

    // ---- layer 1: H = relu(A @ W1 + t*tw1 + b1) ----
    {
        // hoist the 4 A-fragments (invariant across all 8 N-tiles)
        v16h af[4];
#pragma unroll
        for (int kt = 0; kt < 4; kt++) {
            const _Float16* rp = &lds_A[arow][kt * 32 + k0off];
            v8h lo = *(const v8h*)rp;
            v8h hi = *(const v8h*)(rp + 8);
#pragma unroll
            for (int q = 0; q < 8; q++) { af[kt][q] = lo[q]; af[kt][q + 8] = hi[q]; }
        }
#pragma unroll
        for (int nt = 0; nt < 8; nt++) {
            v8f acc = {};
#pragma unroll
            for (int kt = 0; kt < 4; kt++) {
                v16h bm = *(const v16h*)&lds_W[((kt * 8 + nt) * 32 + lane) * 16];
                acc = __builtin_amdgcn_wmma_f32_16x16x32_f16(false, af[kt], false, bm,
                                                             (short)0, acc, false, false);
            }
            int   n  = nt * 16 + nlow;
            float bb = b1[e * D + n];
            float tt = tw1[e * D + n];
#pragma unroll
            for (int j = 0; j < 8; j++) {
                float v = acc[j] + tv[j] * tt + bb;
                v = fmaxf(v, 0.0f);
                int m = (lane < 16) ? j : (8 + j);
                lds_H[16 * w + m][n] = (_Float16)v;
            }
        }
    }
    __syncthreads();
    // ---- stage W2 fragments ----
    {
        const uint4* src = (const uint4*)(W2p + e * (D * D));
        uint4*       dst = (uint4*)lds_W;
#pragma unroll
        for (int c = 0; c < 8; c++) dst[tid + c * 256] = src[tid + c * 256];
    }
    __syncthreads();

    // ---- layer 2: A = relu(H @ W2 + t*tw2 + b2) ----
    {
        v16h af[4];
#pragma unroll
        for (int kt = 0; kt < 4; kt++) {
            const _Float16* rp = &lds_H[arow][kt * 32 + k0off];
            v8h lo = *(const v8h*)rp;
            v8h hi = *(const v8h*)(rp + 8);
#pragma unroll
            for (int q = 0; q < 8; q++) { af[kt][q] = lo[q]; af[kt][q + 8] = hi[q]; }
        }
#pragma unroll
        for (int nt = 0; nt < 8; nt++) {
            v8f acc = {};
#pragma unroll
            for (int kt = 0; kt < 4; kt++) {
                v16h bm = *(const v16h*)&lds_W[((kt * 8 + nt) * 32 + lane) * 16];
                acc = __builtin_amdgcn_wmma_f32_16x16x32_f16(false, af[kt], false, bm,
                                                             (short)0, acc, false, false);
            }
            int   n  = nt * 16 + nlow;
            float bb = b2[e * D + n];
            float tt = tw2[e * D + n];
#pragma unroll
            for (int j = 0; j < 8; j++) {
                float v = acc[j] + tv[j] * tt + bb;
                v = fmaxf(v, 0.0f);
                int m = (lane < 16) ? j : (8 + j);
                lds_A[16 * w + m][n] = (_Float16)v;
            }
        }
    }
    // wave-private rows: no barrier needed (DS ops are in-order per wave)

    // ---- layer 3: out = A @ W3 + t*tw3 + b3 ----
    if (lane < 16) {
        int   row = 16 * w + lane;
        float sum = 0.0f;
#pragma unroll 8
        for (int k = 0; k < D; k++)
            sum += (float)lds_A[row][k] * lds_w3[k];
        int gRow = lds_row[row];
        if (gRow >= 0)
            out[gRow] = sum + lds_t[row] * tw3[e] + b3[e];
    }
}

extern "C" void kernel_launch(void* const* d_in, const int* in_sizes, int n_in,
                              void* d_out, int out_size, void* d_ws, size_t ws_size,
                              hipStream_t stream) {
    const float* x   = (const float*)d_in[0];
    const float* W1  = (const float*)d_in[1];
    const float* tw1 = (const float*)d_in[2];
    const float* b1  = (const float*)d_in[3];
    const float* W2  = (const float*)d_in[4];
    const float* tw2 = (const float*)d_in[5];
    const float* b2  = (const float*)d_in[6];
    const float* W3  = (const float*)d_in[7];
    const float* tw3 = (const float*)d_in[8];
    const float* b3  = (const float*)d_in[9];
    float* out = (float*)d_out;

    char* ws      = (char*)d_ws;
    int*  counts  = (int*)(ws);            // 8 ints
    int*  offsets = (int*)(ws + 32);       // 8 ints
    int*  cursor  = (int*)(ws + 64);       // 8 ints
    int*  idx     = (int*)(ws + 128);      // BROWS ints (1 MB)
    _Float16* W1p = (_Float16*)(ws + 128 + (size_t)BROWS * 4);
    _Float16* W2p = W1p + WSZ;

    hipMemsetAsync(counts, 0, 32, stream);
    hipMemsetAsync(d_out, 0, (size_t)out_size * sizeof(float), stream);

    route_count  <<<BROWS / 256, 256, 0, stream>>>(x, counts);
    route_scan   <<<1, 32, 0, stream>>>(counts, offsets, cursor);
    route_scatter<<<BROWS / 256, 256, 0, stream>>>(x, cursor, idx);
    prep_weights <<<(2 * WSZ) / 256, 256, 0, stream>>>(W1, W2, W1p, W2p);

    moe_mlp<<<NE * NTPE, 256, 0, stream>>>(x, W1p, W2p, tw1, b1, tw2, b2,
                                           W3, tw3, b3, offsets, idx, out);
}
